// NNXAttentionOp_19499151524670
// MI455X (gfx1250) — matrix operand order repaired and checked by reading, not compiled, not run on hardware
//
#include <hip/hip_runtime.h>

// ---------------------------------------------------------------------------
// Flash-attention forward for MI455X (gfx1250, wave32, WMMA f16->f32).
//  Pass 1: convert K -> f16 row-major, V -> f16 *transposed* per head (global
//          scratch). Memory bound, ~6us at 23.3 TB/s.
//  Pass 2: flash attention. KV tiles are moved global->LDS by the Tensor Data
//          Mover (tensor_load_to_lds, TENSORcnt, double buffered, wave 0
//          issues) -- zero loader VALU/VMEM in the loop. Compute: S^T = K*Q^T,
//          O^T = V^T*P^T via v_wmma_f32_16x16x32_f16, softmax per query
//          column held on a lane pair (in-register reductions + 1 xor16).
// ---------------------------------------------------------------------------

typedef __attribute__((ext_vector_type(16))) _Float16 v16h;
typedef __attribute__((ext_vector_type(8)))  _Float16 v8h;
typedef __attribute__((ext_vector_type(4)))  _Float16 v4h;
typedef __attribute__((ext_vector_type(8)))  float    v8f;
typedef __attribute__((ext_vector_type(4)))  float    v4f;
typedef __attribute__((ext_vector_type(4)))  int      v4i;
typedef __attribute__((ext_vector_type(8)))  int      v8i;
typedef __attribute__((ext_vector_type(4)))  unsigned v4u;

#define S_LEN    4096
#define D_MODEL  1024
#define DHEAD    64
#define BH       32
#define BLOCK_M  64
#define BLOCK_N  64
#define NWAVES   4
#define NTILES   (S_LEN / BLOCK_N)
#define LROW     72                 // padded LDS row stride in halves (64+8)

// 1/sqrt(64) * log2(e): fold softmax scale + base-2 conversion into Q.
#define QSCALE   (0.125f * 1.44269504088896341f)

static __device__ inline v16h cat8(v8h lo, v8h hi) {
  return __builtin_shufflevector(lo, hi, 0,1,2,3,4,5,6,7,8,9,10,11,12,13,14,15);
}
static __device__ inline v8h ld8h(const _Float16* p) { return *(const v8h*)p; }
static __device__ inline v4i h2i(v8h x) { return __builtin_bit_cast(v4i, x); }
static __device__ inline v8h i2h(v4i x) { return __builtin_bit_cast(v8h, x); }

static __device__ inline v8f wmma_f16(v16h a, v16h b, v8f c) {
  return __builtin_amdgcn_wmma_f32_16x16x32_f16(false, a, false, b,
                                                (short)0, c, false, false);
}

// Exchange 8 halves (4 dwords) with the other half-wave (lane ^ 16).
static __device__ inline v8h xchg16(v8h x) {
  v4i a = h2i(x);
#pragma unroll
  for (int j = 0; j < 4; ++j) a[j] = __shfl_xor(a[j], 16, 32);
  return i2h(a);
}

// 32-bit LDS byte offset of a __shared__ object (flat LDS addr low 32 bits).
static __device__ inline unsigned lds_off(const void* p) {
  return (unsigned)(reinterpret_cast<uintptr_t>(p));
}

// TDM: DMA a 64x64 f16 tile (rows padded to LROW halves in LDS) from a 2D
// tensor: tensor_d0 x tensor_d1 elements, row stride `stride0` elements.
static __device__ inline void tdm_load_tile(unsigned lds_addr,
                                            unsigned long long gaddr,
                                            unsigned tensor_d0,
                                            unsigned tensor_d1,
                                            unsigned long long stride0) {
  v4u g0;
  g0[0] = 1u;                                       // count=1, user mode
  g0[1] = lds_addr;                                 // LDS byte address
  g0[2] = (unsigned)(gaddr & 0xFFFFFFFFull);        // global_addr[31:0]
  g0[3] = (unsigned)((gaddr >> 32) & 0x1FFFFFFull)  // global_addr[56:32]
          | (2u << 30);                             // type = 2 ("image")
  v8i g1;
  g1[0] = (int)((1u << 16)     // data_size = 1 -> 2 bytes
              | (1u << 20)     // pad_enable
              | (4u << 22)     // pad_interval: every 32 dwords (128B row)
              | (3u << 25));   // pad_amount:   4 dwords (16B)  -> LROW=72
  g1[1] = (int)((tensor_d0 & 0xFFFFu) << 16);                  // dim0[15:0]
  g1[2] = (int)(((tensor_d0 >> 16) & 0xFFFFu)
              | ((tensor_d1 & 0xFFFFu) << 16));                // dim0 hi, dim1 lo
  g1[3] = (int)(((tensor_d1 >> 16) & 0xFFFFu)
              | (64u << 16));                                  // tile_dim0 = 64
  g1[4] = (int)64u;                                            // tile_dim1 = 64
  g1[5] = (int)(unsigned)(stride0 & 0xFFFFFFFFull);            // dim0 stride lo
  g1[6] = (int)(unsigned)((stride0 >> 32) & 0xFFFFull);        // dim0 stride hi
  g1[7] = 0;
  v4i gz4;
  gz4[0] = 0; gz4[1] = 0; gz4[2] = 0; gz4[3] = 0;              // 2D: groups 2/3
  v8i gz8;
#pragma unroll
  for (int i = 0; i < 8; ++i) gz8[i] = 0;
  __builtin_amdgcn_tensor_load_to_lds(g0, g1, gz4, gz4, gz8, 0);
}

// ---------------------------------------------------------------------------
// Pass 1: K -> f16 [bh][key][dh], V -> f16 transposed [bh][dh][key].
// ---------------------------------------------------------------------------
__global__ __launch_bounds__(128, 1)
void preconvert_kernel(const float* __restrict__ k, const float* __restrict__ v,
                       _Float16* __restrict__ kh, _Float16* __restrict__ vt) {
  __shared__ __align__(16) _Float16 sT[DHEAD * LROW];   // V tile transposed

  const int tid = threadIdx.x;
  const int kb  = blockIdx.x;          // key block
  const int bh  = blockIdx.y;
  const int b   = bh >> 4;
  const int h   = bh & 15;
  const int kv0 = kb * BLOCK_N;

#pragma unroll
  for (int it = 0; it < 8; ++it) {
    const int idx = tid + it * 128;    // 0..1023 float4 chunks
    const int row = idx >> 4;
    const int c4  = idx & 15;
    const size_t goff =
        ((size_t)(b * S_LEN + kv0 + row)) * D_MODEL + h * DHEAD + c4 * 4;
    v4f kf = *(const v4f*)(k + goff);
    v4f vf = *(const v4f*)(v + goff);
    v4h kp;
#pragma unroll
    for (int j = 0; j < 4; ++j) {
      kp[j] = (_Float16)kf[j];
      sT[(c4 * 4 + j) * LROW + row] = (_Float16)vf[j];
    }
    *(v4h*)&kh[((size_t)bh * S_LEN + kv0 + row) * DHEAD + c4 * 4] = kp;
  }
  __syncthreads();
  // Write V^T rows (dh-major, keys contiguous) coalesced.
#pragma unroll
  for (int it = 0; it < 4; ++it) {
    const int c   = tid + it * 128;    // 0..511 chunks of 8 halves
    const int row = c >> 3;            // dh
    const int o   = (c & 7) * 8;       // key offset
    v8h x = ld8h(&sT[row * LROW + o]);
    *(v8h*)&vt[((size_t)bh * DHEAD + row) * S_LEN + kv0 + o] = x;
  }
}

// ---------------------------------------------------------------------------
// Pass 2: flash attention, TDM double-buffered KV tiles.
// ---------------------------------------------------------------------------
__global__ __launch_bounds__(128, 1)
void fa_fwd_kernel(const float* __restrict__ q, const _Float16* __restrict__ kh,
                   const _Float16* __restrict__ vt, float* __restrict__ out) {
  __shared__ __align__(16) _Float16 sK [2][BLOCK_N * LROW];  // [key][dh]
  __shared__ __align__(16) _Float16 sVt[2][DHEAD   * LROW];  // [dh][key]

  const int tid  = threadIdx.x;
  const int wave = tid >> 5;
  const int lane = tid & 31;
  const int hw   = lane >> 4;
  const int ln   = lane & 15;
  const bool hi_half = (hw != 0);

  const int mtile = blockIdx.x;
  const int bh    = blockIdx.y;
  const int b     = bh >> 4;
  const int h     = bh & 15;

  const unsigned long long khbase =
      (unsigned long long)(uintptr_t)(kh + (size_t)bh * S_LEN * DHEAD);
  const unsigned long long vtbase =
      (unsigned long long)(uintptr_t)(vt + (size_t)bh * DHEAD * S_LEN);

  // ---- Q -> B-fragments of Q^T (lane = query col ln, K(dh)=16*hw+0..15). --
  const int   qrow = mtile * BLOCK_M + wave * 16 + ln;
  const float* qb  = q + ((size_t)(b * S_LEN + qrow)) * D_MODEL + h * DHEAD + 16 * hw;
  v16h qb0, qb1;
  {
    v4f x0 = *(const v4f*)(qb +  0), x1 = *(const v4f*)(qb +  4);
    v4f x2 = *(const v4f*)(qb +  8), x3 = *(const v4f*)(qb + 12);
    v4f x4 = *(const v4f*)(qb + 32), x5 = *(const v4f*)(qb + 36);
    v4f x6 = *(const v4f*)(qb + 40), x7 = *(const v4f*)(qb + 44);
#pragma unroll
    for (int i = 0; i < 4; ++i) {
      qb0[i]      = (_Float16)(x0[i] * QSCALE);
      qb0[4 + i]  = (_Float16)(x1[i] * QSCALE);
      qb0[8 + i]  = (_Float16)(x2[i] * QSCALE);
      qb0[12 + i] = (_Float16)(x3[i] * QSCALE);
      qb1[i]      = (_Float16)(x4[i] * QSCALE);
      qb1[4 + i]  = (_Float16)(x5[i] * QSCALE);
      qb1[8 + i]  = (_Float16)(x6[i] * QSCALE);
      qb1[12 + i] = (_Float16)(x7[i] * QSCALE);
    }
  }

  v8f zacc;
#pragma unroll
  for (int i = 0; i < 8; ++i) zacc[i] = 0.0f;
  v8f oacc[4];
#pragma unroll
  for (int t = 0; t < 4; ++t) oacc[t] = zacc;
  float m_c = -3.0e38f;
  float l_c = 0.0f;

  // Preload tile 0 (wave 0 only; TDM ignores EXEC, branch keeps it 1 wave).
  if (wave == 0) {
    tdm_load_tile(lds_off(&sK[0][0]),  khbase, DHEAD, S_LEN, DHEAD);
    tdm_load_tile(lds_off(&sVt[0][0]), vtbase, S_LEN, DHEAD, S_LEN);
  }

  for (int nt = 0; nt < NTILES; ++nt) {
    if (wave == 0) __builtin_amdgcn_s_wait_tensorcnt(0);
    __syncthreads();   // tile nt ready; everyone done with buffer (nt+1)&1

    if (wave == 0 && nt + 1 < NTILES) {
      const int nb  = (nt + 1) & 1;
      const unsigned long long kgo = khbase + (unsigned long long)(nt + 1) * BLOCK_N * DHEAD * 2;
      const unsigned long long vgo = vtbase + (unsigned long long)(nt + 1) * BLOCK_N * 2;
      tdm_load_tile(lds_off(&sK[nb][0]),  kgo, DHEAD, S_LEN, DHEAD);
      tdm_load_tile(lds_off(&sVt[nb][0]), vgo, S_LEN, DHEAD, S_LEN);
    }

    const _Float16* kbuf = &sK [nt & 1][0];
    const _Float16* vbuf = &sVt[nt & 1][0];

    // S^T = K Q^T : 4 key tiles x 2 WMMA. C-layout: (key=r+8*hw, query=ln).
    v8f sacc[4];
#pragma unroll
    for (int t = 0; t < 4; ++t) {
      const _Float16* kr = kbuf + (t * 16 + ln) * LROW + 16 * hw;
      v16h ka0 = cat8(ld8h(kr),      ld8h(kr + 8));
      v16h ka1 = cat8(ld8h(kr + 32), ld8h(kr + 40));
      sacc[t] = wmma_f16(ka0, qb0, zacc);
      sacc[t] = wmma_f16(ka1, qb1, sacc[t]);
    }

    // ---- online softmax per query column (base-2 domain) -----------------
    float mx = -3.0e38f;
#pragma unroll
    for (int t = 0; t < 4; ++t)
#pragma unroll
      for (int r = 0; r < 8; ++r) mx = fmaxf(mx, sacc[t][r]);
    mx = fmaxf(mx, __shfl_xor(mx, 16, 32));

    const float mn    = fmaxf(m_c, mx);
    const float alpha = exp2f(m_c - mn);
    m_c = mn;

    float sm = 0.0f;
    v8h pk[4];
#pragma unroll
    for (int t = 0; t < 4; ++t) {
#pragma unroll
      for (int r = 0; r < 8; ++r) {
        const float pv = exp2f(sacc[t][r] - mn);
        sm += pv;
        pk[t][r] = (_Float16)pv;
      }
    }
    sm += __shfl_xor(sm, 16, 32);
    l_c = l_c * alpha + sm;
#pragma unroll
    for (int t = 0; t < 4; ++t)
#pragma unroll
      for (int r = 0; r < 8; ++r) oacc[t][r] *= alpha;

    // ---- P^T B-fragments: halves j = key 16*hw + j of this lane's column.
    v8h xk0 = xchg16(pk[0]);
    v8h xk1 = xchg16(pk[1]);
    v8h xk2 = xchg16(pk[2]);
    v8h xk3 = xchg16(pk[3]);
    v8h lo0 = hi_half ? xk1 : pk[0];
    v8h hi0 = hi_half ? pk[1] : xk0;
    v8h lo1 = hi_half ? xk3 : pk[2];
    v8h hi1 = hi_half ? pk[3] : xk2;
    v16h pb0 = cat8(lo0, hi0);   // keys  0..31
    v16h pb1 = cat8(lo1, hi1);   // keys 32..63

    // O^T += V^T P^T : 4 dh tiles x 2 WMMA.
#pragma unroll
    for (int t = 0; t < 4; ++t) {
      const _Float16* vr = vbuf + (t * 16 + ln) * LROW + 16 * hw;
      v16h va0 = cat8(ld8h(vr),      ld8h(vr + 8));
      v16h va1 = cat8(ld8h(vr + 32), ld8h(vr + 40));
      oacc[t] = wmma_f16(va0, pb0, oacc[t]);
      oacc[t] = wmma_f16(va1, pb1, oacc[t]);
    }
  }

  // ---- epilogue: O^T / l --------------------------------------------------
  const float invl = 1.0f / l_c;
  float* ob = out + ((size_t)bh * S_LEN + qrow) * DHEAD;
#pragma unroll
  for (int t = 0; t < 4; ++t) {
    v4f s0, s1;
#pragma unroll
    for (int j = 0; j < 4; ++j) {
      s0[j] = oacc[t][j]     * invl;
      s1[j] = oacc[t][4 + j] * invl;
    }
    float* op = ob + t * 16 + 8 * hw;
    *(v4f*)(op)     = s0;
    *(v4f*)(op + 4) = s1;
  }
}

extern "C" void kernel_launch(void* const* d_in, const int* in_sizes, int n_in,
                              void* d_out, int out_size, void* d_ws, size_t ws_size,
                              hipStream_t stream) {
  (void)in_sizes; (void)n_in; (void)out_size; (void)ws_size;
  const float* q = (const float*)d_in[0];
  const float* k = (const float*)d_in[1];
  const float* v = (const float*)d_in[2];
  float* out = (float*)d_out;

  _Float16* kh = (_Float16*)d_ws;                                  // 16 MB
  _Float16* vt = (_Float16*)((char*)d_ws + (size_t)BH * S_LEN * DHEAD * 2);

  dim3 grid(NTILES, BH);
  dim3 block(32 * NWAVES);
  preconvert_kernel<<<grid, block, 0, stream>>>(k, v, kh, vt);
  fa_fwd_kernel<<<grid, block, 0, stream>>>(q, kh, vt, out);
}
